// SigLIPModel_22505628631375
// MI455X (gfx1250) — compile-verified, matrix operand order checked by reading, not run on hardware
//
#include <hip/hip_runtime.h>
#include <hip/hip_bf16.h>

#define NROWS 8192
#define DDIM  1024

typedef __attribute__((ext_vector_type(16))) _Float16 v16h;
typedef __attribute__((ext_vector_type(8)))  _Float16 v8h;
typedef __attribute__((ext_vector_type(4)))  _Float16 v4h;
typedef __attribute__((ext_vector_type(8)))  float    v8f;

// ---------------------------------------------------------------------------
// Kernel 1: L2-normalize rows of both embedding matrices, store as f16.
// One wave32 per row; 8 rows per 256-thread block. 16384 rows total.
// ---------------------------------------------------------------------------
__global__ __launch_bounds__(256) void normalize_f16(
    const float* __restrict__ img, const float* __restrict__ txt,
    _Float16* __restrict__ imgH, _Float16* __restrict__ txtH)
{
    const int row  = blockIdx.x * 8 + (threadIdx.x >> 5);
    const int lane = threadIdx.x & 31;

    const float* __restrict__ src;
    _Float16*    __restrict__ dst;
    if (row < NROWS) {
        src = img + (size_t)row * DDIM;
        dst = imgH + (size_t)row * DDIM;
    } else {
        src = txt + (size_t)(row - NROWS) * DDIM;
        dst = txtH + (size_t)(row - NROWS) * DDIM;
    }

    // Each lane: 8 float4 loads, strided -> coalesced b128 loads.
    float4 v[8];
    float ss = 0.0f;
#pragma unroll
    for (int i = 0; i < 8; ++i) {
        v[i] = ((const float4*)src)[lane + i * 32];
        ss += v[i].x * v[i].x + v[i].y * v[i].y + v[i].z * v[i].z + v[i].w * v[i].w;
    }
    // wave32 butterfly reduction
#pragma unroll
    for (int off = 16; off > 0; off >>= 1)
        ss += __shfl_xor(ss, off, 32);

    const float rn = rsqrtf(ss + 1e-12f);

#pragma unroll
    for (int i = 0; i < 8; ++i) {
        v4h h;
        h[0] = (_Float16)(v[i].x * rn);
        h[1] = (_Float16)(v[i].y * rn);
        h[2] = (_Float16)(v[i].z * rn);
        h[3] = (_Float16)(v[i].w * rn);
        ((v4h*)dst)[lane + i * 32] = h;   // 8B coalesced stores
    }
}

// ---------------------------------------------------------------------------
// Fragment load per CDNA5 16-bit A/B WMMA layout (16x32 / 32x16):
//   lanes 0-15 : M/N = lane,    K = k0+0..7   (v0..v3) and k0+16..23 (v4..v7)
//   lanes16-31 : M/N = lane-16, K = k0+8..15  and k0+24..31
// Both operands contract along memory rows (einsum "id,jd->ij"), so the same
// loader serves A (image rows) and B (text rows).
// ---------------------------------------------------------------------------
__device__ __forceinline__ v16h load_frag(const _Float16* __restrict__ base,
                                          int rowBase, int k0, int lane)
{
    const int r  = rowBase + (lane & 15);
    const int kb = k0 + ((lane >> 4) << 3);
    const _Float16* p = base + (size_t)r * DDIM + kb;
    v8h lo = *(const v8h*)(p);        // K = kb .. kb+7
    v8h hi = *(const v8h*)(p + 16);   // K = kb+16 .. kb+23
    return __builtin_shufflevector(lo, hi, 0, 1, 2, 3, 4, 5, 6, 7,
                                           8, 9, 10, 11, 12, 13, 14, 15);
}

__device__ __forceinline__ float log_sigmoid(float s)
{
    // stable: min(s,0) - log1p(exp(-|s|))
    return fminf(s, 0.0f) - log1pf(expf(-fabsf(s)));
}

// ---------------------------------------------------------------------------
// Kernel 2: fused GEMM (f16 WMMA, f32 acc) + SigLIP log-sigmoid reduction.
// Block = 256 threads = 8 waves arranged 2 (M) x 4 (N); block tile 128x256.
// Wave tile 64x64 = 4x4 WMMA accumulators -> 16 WMMAs per 8 fragment loads
// per K-step (32 FLOP/B from L2). Deterministic per-block partial sums.
// ---------------------------------------------------------------------------
__global__ __launch_bounds__(256) void gemm_loss(
    const _Float16* __restrict__ imgH, const _Float16* __restrict__ txtH,
    const float* __restrict__ tptr, const float* __restrict__ bptr,
    float* __restrict__ partials)
{
    const int tid    = threadIdx.x;
    const int lane   = tid & 31;
    const int wave   = tid >> 5;
    const int wm     = wave >> 2;   // 0..1  (64-row slabs)
    const int wn     = wave & 3;    // 0..3  (64-col slabs)
    const int lane16 = lane & 15;
    const int laneHi = lane >> 4;

    const int m0 = blockIdx.x * 128 + wm * 64;
    const int n0 = blockIdx.y * 256 + wn * 64;

    v8f acc[4][4];
#pragma unroll
    for (int i = 0; i < 4; ++i)
#pragma unroll
        for (int j = 0; j < 4; ++j)
            acc[i][j] = (v8f){0.f, 0.f, 0.f, 0.f, 0.f, 0.f, 0.f, 0.f};

#pragma unroll 2
    for (int k0 = 0; k0 < DDIM; k0 += 32) {
        v16h a[4], b[4];
#pragma unroll
        for (int i = 0; i < 4; ++i) a[i] = load_frag(imgH, m0 + i * 16, k0, lane);
#pragma unroll
        for (int j = 0; j < 4; ++j) b[j] = load_frag(txtH, n0 + j * 16, k0, lane);

#pragma unroll
        for (int i = 0; i < 4; ++i)
#pragma unroll
            for (int j = 0; j < 4; ++j)
                acc[i][j] = __builtin_amdgcn_wmma_f32_16x16x32_f16(
                    /*neg_a=*/false, a[i], /*neg_b=*/false, b[j],
                    /*c_mod=*/(short)0, acc[i][j],
                    /*reuse_a=*/false, /*reuse_b=*/false);
    }

    const float expT = expf(tptr[0]);
    const float bias = bptr[0];

    // C/D layout: VGPR e -> M = e (lanes 0-15) or e+8 (lanes 16-31); N = lane&15.
    float local = 0.0f;
#pragma unroll
    for (int i = 0; i < 4; ++i) {
#pragma unroll
        for (int j = 0; j < 4; ++j) {
#pragma unroll
            for (int e = 0; e < 8; ++e) {
                const int row = m0 + i * 16 + e + laneHi * 8;
                const int col = n0 + j * 16 + lane16;
                const float lg = acc[i][j][e] * expT + bias;
                const float s  = (row == col) ? lg : -lg;
                local += log_sigmoid(s);
            }
        }
    }

    // Deterministic block reduction (fixed tree shape).
    __shared__ float red[256];
    red[tid] = local;
    __syncthreads();
#pragma unroll
    for (int s = 128; s > 0; s >>= 1) {
        if (tid < s) red[tid] += red[tid + s];
        __syncthreads();
    }
    if (tid == 0)
        partials[blockIdx.y * gridDim.x + blockIdx.x] = red[0];
}

// ---------------------------------------------------------------------------
// Kernel 3: deterministic final reduction of 2048 partials -> loss scalar.
// ---------------------------------------------------------------------------
__global__ __launch_bounds__(256) void finish(
    const float* __restrict__ partials, float* __restrict__ out)
{
    __shared__ float red[256];
    float s = 0.0f;
    for (int i = threadIdx.x; i < 2048; i += 256) s += partials[i]; // fixed order
    red[threadIdx.x] = s;
    __syncthreads();
#pragma unroll
    for (int k = 128; k > 0; k >>= 1) {
        if (threadIdx.x < k) red[threadIdx.x] += red[threadIdx.x + k];
        __syncthreads();
    }
    if (threadIdx.x == 0)
        out[0] = -red[0] / (float)NROWS;
}

extern "C" void kernel_launch(void* const* d_in, const int* in_sizes, int n_in,
                              void* d_out, int out_size, void* d_ws, size_t ws_size,
                              hipStream_t stream)
{
    const float* img  = (const float*)d_in[0];
    const float* txt  = (const float*)d_in[1];
    const float* temp = (const float*)d_in[2];
    const float* bias = (const float*)d_in[3];
    float* out = (float*)d_out;

    const size_t matBytes = (size_t)NROWS * DDIM * sizeof(_Float16); // 16 MiB
    char* ws = (char*)d_ws;
    _Float16* imgH     = (_Float16*)(ws);
    _Float16* txtH     = (_Float16*)(ws + matBytes);
    float*    partials = (float*)(ws + 2 * matBytes);

    // 16384 rows, 8 rows (waves) per block
    normalize_f16<<<2048, 256, 0, stream>>>(img, txt, imgH, txtH);

    dim3 grid(NROWS / 128, NROWS / 256);   // 64 x 32 blocks, tile 128x256
    gemm_loss<<<grid, 256, 0, stream>>>(imgH, txtH, temp, bias, partials);

    finish<<<1, 256, 0, stream>>>(partials, out);
}
